// GQA_34050500723478
// MI455X (gfx1250) — compile-verified, matrix operand order checked by reading, not run on hardware
//
#include <hip/hip_runtime.h>
#include <hip/hip_bf16.h>

// ---------------------------------------------------------------------------
// GQA block for MI455X (gfx1250, wave32, WMMA + TDM async tensor loads).
//   B=2 S=2048 D=2048 H=16 G=4 HD=128 KV=512
// Pipeline: cast fp32->bf16, QKV GEMM (bf16 WMMA, f32 acc, TDM-staged W),
// flash attention (TDM-staged K/V chunks, bf16 WMMA scores + PV, f32 online
// softmax), out GEMM (bf16 WMMA, f32 out).
// ---------------------------------------------------------------------------

#define GQA_B   2
#define GQA_S   2048
#define GQA_D   2048
#define GQA_H   16
#define GQA_G   4
#define GQA_HD  128
#define GQA_KV  512
#define GQA_QKVW (GQA_D + 2 * GQA_KV)   // 3072

typedef __attribute__((ext_vector_type(16))) __bf16 v16bf;
typedef __attribute__((ext_vector_type(8)))  __bf16 v8bf;
typedef __attribute__((ext_vector_type(8)))  float  v8f;
typedef __attribute__((ext_vector_type(4)))  unsigned u32x4;
typedef __attribute__((ext_vector_type(8)))  int      i32x8;
typedef __attribute__((ext_vector_type(4)))  int      i32x4;

#if defined(__HIP_DEVICE_COMPILE__) && \
    __has_builtin(__builtin_amdgcn_tensor_load_to_lds) && \
    __has_builtin(__builtin_amdgcn_s_wait_tensorcnt)
#define GQA_TDM 1
#else
#define GQA_TDM 0
#endif

// --- WMMA wrapper: D = A(16x32 bf16) * B(32x16 bf16) + C(16x16 f32) ---------
__device__ __forceinline__ v8f wmma_bf16(v16bf a, v16bf b, v8f c) {
    return __builtin_amdgcn_wmma_f32_16x16x32_bf16(
        /*neg_a=*/false, a, /*neg_b=*/false, b,
        /*c_mod=*/(short)0, c, /*reuse_a=*/false, /*reuse_b=*/false);
}

// A-fragment (16x32 tile, 16-bit): lane r=lane&15 owns row r; elements
// e=0..7 hold K = 8*hi + e, e=8..15 hold K = 16 + 8*hi + (e-8). Both runs are
// contiguous 16-byte chunks -> two vector loads (global_load_b128/ds_load_b128).
__device__ __forceinline__ v16bf load_fragA(const __bf16* row_k0, int hi) {
    v8bf lo  = *reinterpret_cast<const v8bf*>(row_k0 + 8 * hi);
    v8bf hi8 = *reinterpret_cast<const v8bf*>(row_k0 + 16 + 8 * hi);
    return __builtin_shufflevector(lo, hi8,
        0, 1, 2, 3, 4, 5, 6, 7, 8, 9, 10, 11, 12, 13, 14, 15);
}

// --- TDM: async 2D bf16 tile load (global -> LDS), D# per ISA 08 §8 ---------
// Tile declared as its own tensor (tile origin == tensor origin, never OOB).
__device__ __forceinline__ void tdm_load_tile(__bf16* lds_dst, const __bf16* gsrc,
                                              unsigned stride_elems,
                                              unsigned tw, unsigned th) {
#if GQA_TDM
    unsigned lds_addr = (unsigned)(unsigned long long)(size_t)lds_dst;
    unsigned long long ga = (unsigned long long)(size_t)gsrc;
    u32x4 g0;
    g0[0] = 1u;                                       // count=1, user mode
    g0[1] = lds_addr;                                 // lds_addr (bytes)
    g0[2] = (unsigned)(ga & 0xFFFFFFFFu);             // global_addr[31:0]
    g0[3] = (unsigned)((ga >> 32) & 0x01FFFFFFu) | (2u << 30);  // addr hi | type=2
    i32x8 g1;
    g1[0] = (int)(1u << 16);                          // data_size=1 (2 bytes)
    g1[1] = (int)((tw & 0xFFFFu) << 16);              // tensor_dim0[15:0]
    g1[2] = (int)(((tw >> 16) & 0xFFFFu) | ((th & 0xFFFFu) << 16));   // dim0 hi | dim1 lo
    g1[3] = (int)(((th >> 16) & 0xFFFFu) | ((tw & 0xFFFFu) << 16));   // dim1 hi | tile_dim0
    g1[4] = (int)(th & 0xFFFFu);                      // tile_dim1 (tile_dim2=0)
    g1[5] = (int)stride_elems;                        // tensor_dim0_stride[31:0]
    g1[6] = 0;                                        // stride hi | dim1_stride lo
    g1[7] = 0;
    i32x4 z4 = {};
#if __clang_major__ >= 23
    i32x8 z8 = {};
    __builtin_amdgcn_tensor_load_to_lds(g0, g1, z4, z4, z8, 0);
#else
    __builtin_amdgcn_tensor_load_to_lds(g0, g1, z4, z4, 0);
#endif
#else
    // Fallback: cooperative synchronous vector copy (one wave).
    const int lane = threadIdx.x & 31;
    const unsigned cpr = tw / 8;            // v8bf chunks per row
    for (unsigned i = lane; i < th * cpr; i += 32) {
        unsigned row = i / cpr, c = i % cpr;
        *reinterpret_cast<v8bf*>(lds_dst + (size_t)row * tw + c * 8) =
            *reinterpret_cast<const v8bf*>(gsrc + (size_t)row * stride_elems + c * 8);
    }
#endif
}

// Wait until at most `pending` TDM ops outstanding, then make LDS visible.
__device__ __forceinline__ void tdm_wait0() {
#if GQA_TDM
    __builtin_amdgcn_s_wait_tensorcnt(0);
#endif
    __syncthreads();
}
__device__ __forceinline__ void tdm_wait1() {
#if GQA_TDM
    __builtin_amdgcn_s_wait_tensorcnt(1);
#endif
    __syncthreads();
}
__device__ __forceinline__ void tdm_wait2() {
#if GQA_TDM
    __builtin_amdgcn_s_wait_tensorcnt(2);
#endif
    __syncthreads();
}

// --- fp32 -> bf16 cast ------------------------------------------------------
__global__ __launch_bounds__(256) void gqa_cast_bf16(const float* __restrict__ in,
                                                     __bf16* __restrict__ out, long n) {
    long i = (long)blockIdx.x * blockDim.x + threadIdx.x;
    long stride = (long)gridDim.x * blockDim.x;
    for (; i < n; i += stride) out[i] = (__bf16)in[i];
}

// --- GEMM: C[M,N] = A[M,K](bf16) @ W[K,N](bf16) + bias[N](f32) --------------
// One wave per 16x64 strip. W tiles (32x64) are TDM-staged into LDS,
// double-buffered on TENSORcnt; B-fragments gathered from LDS.
template <bool OUT_BF16>
__global__ __launch_bounds__(32, 1)
void gqa_gemm_wmma(const __bf16* __restrict__ A,
                   const __bf16* __restrict__ W,
                   const float* __restrict__ bias,
                   void* __restrict__ out,
                   int M, int N, int K) {
    const int lane = threadIdx.x;
    const int r = lane & 15, hi = lane >> 4;
    const int m0 = blockIdx.y * 16;
    const int n0 = blockIdx.x * 64;

    __shared__ __align__(16) __bf16 Wlds[2][32 * 64];   // 8 KB double buffer

    v8f acc[4] = {v8f{}, v8f{}, v8f{}, v8f{}};
    const __bf16* arow = A + (size_t)(m0 + r) * K;

    tdm_load_tile(Wlds[0], W + (size_t)0 * N + n0, (unsigned)N, 64, 32);

    const int nk = K / 32;
    for (int kt = 0; kt < nk; ++kt) {
        const int k0 = kt * 32;
        if (kt + 1 < nk) {
            tdm_load_tile(Wlds[(kt + 1) & 1], W + (size_t)(k0 + 32) * N + n0,
                          (unsigned)N, 64, 32);
            tdm_wait1();                 // current tile landed
        } else {
            tdm_wait0();
        }
        const __bf16* wt = Wlds[kt & 1];

        v16bf af = load_fragA(arow + k0, hi);
        if (k0 + 32 < K)
            __builtin_prefetch(arow + k0 + 32, 0, 1);   // global_prefetch_b8
#pragma unroll
        for (int t = 0; t < 4; ++t) {
            v16bf bf;
#pragma unroll
            for (int e = 0; e < 16; ++e) {
                int kk = e + ((e >= 8) ? 8 : 0) + 8 * hi;
                bf[e] = wt[kk * 64 + t * 16 + r];
            }
            acc[t] = wmma_bf16(af, bf, acc[t]);
        }
        __syncthreads();   // done reading this buffer before TDM overwrites it
    }

#pragma unroll
    for (int t = 0; t < 4; ++t) {
#pragma unroll
        for (int v = 0; v < 8; ++v) {
            int row = m0 + v + 8 * hi;
            int col = n0 + t * 16 + r;
            float val = acc[t][v] + bias[col];
            if (OUT_BF16)
                ((__bf16*)out)[(size_t)row * N + col] = (__bf16)val;
            else
                ((float*)out)[(size_t)row * N + col] = val;
        }
    }
}

// --- Flash attention: one wave per (b, h, 16-query tile) --------------------
// qkv: [B, S, 3072] bf16 (q | k | v). K/V 32x128 chunks TDM-staged into LDS,
// double-buffered. attn_out: [B, S, D] bf16, col = h*HD+hd.
__global__ __launch_bounds__(32, 1)
void gqa_attn_wmma(const __bf16* __restrict__ qkv,
                   __bf16* __restrict__ attn_out) {
    const int lane = threadIdx.x;
    const int r = lane & 15, hi = lane >> 4;
    const int qt = blockIdx.x;          // query tile (S/16)
    const int h  = blockIdx.y;          // head
    const int b  = blockIdx.z;          // batch
    const int g  = h % GQA_G;           // tile() => head h reads group h % G

    const size_t qs = (size_t)GQA_QKVW;                    // qkv row stride
    const __bf16* base  = qkv + (size_t)b * GQA_S * qs;
    const __bf16* qrow  = base + (size_t)(qt * 16 + r) * qs + h * GQA_HD;
    const __bf16* kbase = base + GQA_D + g * GQA_HD;           // + key*qs
    const __bf16* vbase = base + GQA_D + GQA_KV + g * GQA_HD;  // + key*qs

    __shared__ __align__(16) __bf16 Klds[2][32 * GQA_HD];   // 2 x 8 KB
    __shared__ __align__(16) __bf16 Vlds[2][32 * GQA_HD];   // 2 x 8 KB
    __shared__ __align__(16) __bf16 Plds[16 * 32];          // 1 KB

    // Q fragments: HD=128 -> 4 chunks of K=32, resident in registers.
    v16bf qf[4];
#pragma unroll
    for (int kc = 0; kc < 4; ++kc) qf[kc] = load_fragA(qrow + kc * 32, hi);

    v8f acc_o[8];
    float mrow[8], lrow[8];
#pragma unroll
    for (int ct = 0; ct < 8; ++ct) acc_o[ct] = v8f{};
#pragma unroll
    for (int v = 0; v < 8; ++v) { mrow[v] = -1e30f; lrow[v] = 0.0f; }

    const float scale = 1.0f / (float)GQA_HD;        // reference: /HD
    const int NB = GQA_S / 32;                       // 64 key chunks

    tdm_load_tile(Klds[0], kbase, (unsigned)qs, GQA_HD, 32);
    tdm_load_tile(Vlds[0], vbase, (unsigned)qs, GQA_HD, 32);

    for (int kb = 0; kb < NB; ++kb) {
        if (kb + 1 < NB) {
            const size_t roff = (size_t)(kb + 1) * 32 * qs;
            tdm_load_tile(Klds[(kb + 1) & 1], kbase + roff, (unsigned)qs, GQA_HD, 32);
            tdm_load_tile(Vlds[(kb + 1) & 1], vbase + roff, (unsigned)qs, GQA_HD, 32);
            tdm_wait2();                 // current K/V pair landed
        } else {
            tdm_wait0();
        }
        const __bf16* kt = Klds[kb & 1];
        const __bf16* vt = Vlds[kb & 1];

        // ---- scores: two 16x16 tiles over this 32-key chunk ----
        v8f sc[2] = {v8f{}, v8f{}};
#pragma unroll
        for (int t = 0; t < 2; ++t) {
#pragma unroll
            for (int kc = 0; kc < 4; ++kc) {
                // B = K^T: lane = key column (row in Klds), hd contiguous.
                v16bf kf = load_fragA(kt + (t * 16 + r) * GQA_HD + kc * 32, hi);
                sc[t] = wmma_bf16(qf[kc], kf, sc[t]);
            }
        }

        // ---- online softmax (f32), rows = v + 8*hi, cols = lane&15 ----
        float p0[8], p1[8];
#pragma unroll
        for (int v = 0; v < 8; ++v) {
            float s0 = sc[0][v] * scale;
            float s1 = sc[1][v] * scale;
            float mx = fmaxf(s0, s1);
#pragma unroll
            for (int m = 8; m >= 1; m >>= 1) mx = fmaxf(mx, __shfl_xor(mx, m, 32));
            float mnew = fmaxf(mrow[v], mx);
            float corr = __expf(mrow[v] - mnew);
            float e0 = __expf(s0 - mnew);
            float e1 = __expf(s1 - mnew);
            float ps = e0 + e1;
#pragma unroll
            for (int m = 8; m >= 1; m >>= 1) ps += __shfl_xor(ps, m, 32);
            lrow[v] = lrow[v] * corr + ps;
            mrow[v] = mnew;
#pragma unroll
            for (int ct = 0; ct < 8; ++ct) acc_o[ct][v] *= corr;
            p0[v] = e0; p1[v] = e1;
        }

        // ---- transpose P (C-layout -> A-layout) through LDS ----
        __syncthreads();   // prior P reads done before overwrite
#pragma unroll
        for (int v = 0; v < 8; ++v) {
            Plds[(v + 8 * hi) * 32 + r]      = (__bf16)p0[v];
            Plds[(v + 8 * hi) * 32 + 16 + r] = (__bf16)p1[v];
        }
        __syncthreads();   // stores visible before A-fragment loads
        v16bf pf = load_fragA(&Plds[r * 32], hi);

        // ---- O += P(16x32) @ V(32x16) for 8 HD column tiles (LDS gather) ----
#pragma unroll
        for (int ct = 0; ct < 8; ++ct) {
            v16bf vf;
#pragma unroll
            for (int e = 0; e < 16; ++e) {
                int kk = e + ((e >= 8) ? 8 : 0) + 8 * hi;
                vf[e] = vt[kk * GQA_HD + ct * 16 + r];
            }
            acc_o[ct] = wmma_bf16(pf, vf, acc_o[ct]);
        }
        __syncthreads();   // done reading this K/V buffer before TDM overwrite
    }

    // ---- epilogue: O /= l, write bf16 [B,S,D] with col = h*HD + hd ----
    __bf16* orow = attn_out + (size_t)b * GQA_S * GQA_D + (size_t)h * GQA_HD;
#pragma unroll
    for (int ct = 0; ct < 8; ++ct) {
#pragma unroll
        for (int v = 0; v < 8; ++v) {
            float o = acc_o[ct][v] / lrow[v];
            size_t row = (size_t)(qt * 16 + v + 8 * hi);
            orow[row * GQA_D + ct * 16 + r] = (__bf16)o;
        }
    }
}

// ---------------------------------------------------------------------------
extern "C" void kernel_launch(void* const* d_in, const int* in_sizes, int n_in,
                              void* d_out, int out_size, void* d_ws, size_t ws_size,
                              hipStream_t stream) {
    const float* x    = (const float*)d_in[0];   // [B,S,D]
    const float* Wqkv = (const float*)d_in[1];   // [D, D+2KV]
    const float* bqkv = (const float*)d_in[2];   // [D+2KV]
    const float* Wout = (const float*)d_in[3];   // [D, D]
    const float* bout = (const float*)d_in[4];   // [D]
    float* out = (float*)d_out;                  // [B,S,D]

    const long nx    = (long)GQA_B * GQA_S * GQA_D;          // 8,388,608
    const long nwqkv = (long)GQA_D * GQA_QKVW;               // 6,291,456
    const long nwout = (long)GQA_D * GQA_D;                  // 4,194,304
    const long nqkv  = (long)GQA_B * GQA_S * GQA_QKVW;       // 12,582,912
    const long nattn = (long)GQA_B * GQA_S * GQA_D;          // 8,388,608

    char* ws = (char*)d_ws;
    size_t off = 0;
    __bf16* x_bf    = (__bf16*)(ws + off); off += (size_t)nx    * 2;
    __bf16* wqkv_bf = (__bf16*)(ws + off); off += (size_t)nwqkv * 2;
    __bf16* wout_bf = (__bf16*)(ws + off); off += (size_t)nwout * 2;
    __bf16* qkv_bf  = (__bf16*)(ws + off); off += (size_t)nqkv  * 2;
    __bf16* attn_bf = (__bf16*)(ws + off); off += (size_t)nattn * 2;
    (void)ws_size; (void)in_sizes; (void)n_in; (void)out_size;

    // Stage 0: casts to bf16.
    gqa_cast_bf16<<<4096, 256, 0, stream>>>(x,    x_bf,    nx);
    gqa_cast_bf16<<<4096, 256, 0, stream>>>(Wqkv, wqkv_bf, nwqkv);
    gqa_cast_bf16<<<4096, 256, 0, stream>>>(Wout, wout_bf, nwout);

    const int M = GQA_B * GQA_S;   // 4096 rows

    // Stage 1: qkv = x @ Wqkv + bqkv  -> bf16 [M, 3072]
    {
        dim3 grid(GQA_QKVW / 64, M / 16);
        gqa_gemm_wmma<true><<<grid, 32, 0, stream>>>(
            x_bf, wqkv_bf, bqkv, (void*)qkv_bf, M, GQA_QKVW, GQA_D);
    }

    // Stage 2: flash attention -> bf16 [M, D]
    {
        dim3 grid(GQA_S / 16, GQA_H, GQA_B);
        gqa_attn_wmma<<<grid, 32, 0, stream>>>(qkv_bf, attn_bf);
    }

    // Stage 3: out = attn @ Wout + bout -> f32 [M, D]
    {
        dim3 grid(GQA_D / 64, M / 16);
        gqa_gemm_wmma<false><<<grid, 32, 0, stream>>>(
            attn_bf, wout_bf, bout, (void*)out, M, GQA_D, GQA_D);
    }
}
